// Attention_52072183497065
// MI455X (gfx1250) — matrix operand order repaired and checked by reading, not compiled, not run on hardware
//
#include <hip/hip_runtime.h>
#include <hip/hip_bf16.h>

typedef __bf16 bf16_t;
typedef __attribute__((ext_vector_type(16))) __bf16 v16bf;
typedef __attribute__((ext_vector_type(8)))  __bf16 v8bf;
typedef __attribute__((ext_vector_type(8)))  float  v8f;
typedef __attribute__((ext_vector_type(4)))  int    v4i;
typedef __attribute__((ext_vector_type(4)))  unsigned int u32x4;
typedef __attribute__((ext_vector_type(8)))  int    i32x8;
typedef __attribute__((ext_vector_type(4)))  int    i32x4;

#define B_   2
#define N_   2048
#define D_   1024
#define H_   16
#define DH_  64
#define F_   1024          // H_*DH_
#define SCALE_ 0.125f      // DH^-0.5
#define CLAMP_ 50.0f

// ---------------------------------------------------------------------------
// CDNA5 async global->LDS copy (16B per lane) — signature probed in round 2.
// ---------------------------------------------------------------------------
#if defined(__gfx1250__) && __has_builtin(__builtin_amdgcn_global_load_async_to_lds_b128)
#define HAVE_ASYNC_LDS 1
#else
#define HAVE_ASYNC_LDS 0
#endif

typedef __attribute__((address_space(1))) v4i as1_v4i;
typedef __attribute__((address_space(3))) v4i as3_v4i;

__device__ __forceinline__ void async_copy16(bf16_t* ldsp, const bf16_t* gp) {
#if HAVE_ASYNC_LDS
  __builtin_amdgcn_global_load_async_to_lds_b128(
      (as1_v4i*)(size_t)gp, (as3_v4i*)(size_t)ldsp, 0, 0);
#else
  *(v8bf*)ldsp = *(const v8bf*)gp;
#endif
}

__device__ __forceinline__ void async_copy_wait() {
#if HAVE_ASYNC_LDS
  asm volatile("s_wait_asynccnt 0" ::: "memory");
#endif
}

// ---------------------------------------------------------------------------
// CDNA5 Tensor Data Mover: 2D tile load global->LDS (D# per ISA 8.3/8.4).
// Toolchain probe (round 4 diagnostic): 6-arg builtin
//   (u32x4 g0, i32x8 g1, i32x4 g2, i32x4 g3, i32x8 extra, i32 cpol)
// ---------------------------------------------------------------------------
#if defined(__gfx1250__) && __has_builtin(__builtin_amdgcn_tensor_load_to_lds)
#define HAVE_TDM 1
#else
#define HAVE_TDM 0
#endif

// tile_d0 elems along contiguous dim, tile_d1 lines; stride0 = line stride
// (elements); element size fixed at 2 bytes (bf16).  LDS receives the tile
// densely packed (tile_d1 x tile_d0 row-major).
__device__ __forceinline__ void tdm_load_2d(bf16_t* ldsp, const bf16_t* gp,
                                            unsigned tile_d0, unsigned tile_d1,
                                            unsigned stride0) {
#if HAVE_TDM
  size_t ga = (size_t)gp;
  u32x4 g0;
  g0[0] = 1u;                                   // count=1 (valid descriptor)
  g0[1] = (unsigned)(size_t)ldsp;               // lds_addr (flat low 32 = LDS)
  g0[2] = (unsigned)ga;                         // global_addr[31:0]
  g0[3] = (unsigned)((ga >> 32) & 0x01FFFFFFu)  // global_addr[56:32]
        | (2u << 30);                           // type = 2 ("image")
  i32x8 g1;
  g1[0] = (int)(1u << 16);                      // data_size=1 (2 bytes)
  g1[1] = (int)(tile_d0 << 16);                 // tensor_dim0[15:0] @bit48
  g1[2] = (int)((tile_d0 >> 16) | (tile_d1 << 16));   // dim0 hi | tensor_dim1 lo
  g1[3] = (int)((tile_d1 >> 16) | (tile_d0 << 16));   // dim1 hi | tile_dim0
  g1[4] = (int)(tile_d1 & 0xFFFFu);             // tile_dim1 (tile_dim2 = 0)
  g1[5] = (int)stride0;                         // tensor_dim0_stride[31:0]
  g1[6] = 0;                                    // stride hi | dim1_stride lo
  g1[7] = 0;
  i32x4 z4 = {0, 0, 0, 0};
  i32x8 z8 = {0, 0, 0, 0, 0, 0, 0, 0};
  __builtin_amdgcn_tensor_load_to_lds(g0, g1, z4, z4, z8, 0);
#else
  (void)ldsp; (void)gp; (void)tile_d0; (void)tile_d1; (void)stride0;
#endif
}

__device__ __forceinline__ void tdm_wait() {
#if HAVE_TDM
#if __has_builtin(__builtin_amdgcn_s_wait_tensorcnt)
  __builtin_amdgcn_s_wait_tensorcnt((short)0);
#else
  asm volatile("s_wait_tensorcnt 0" ::: "memory");
#endif
#endif
}

// ---------------------------------------------------------------------------
// WMMA helper: D = A(16x32 bf16) * B(32x16 bf16) + C(16x16 f32)
// ---------------------------------------------------------------------------
__device__ __forceinline__ v8f wmma_bf16(v16bf a, v16bf b, v8f c) {
  return __builtin_amdgcn_wmma_f32_16x16x32_bf16(
      false, a, false, b, (short)0, c, false, false);
}

// A-fragment (16x32, MxK), row-major source, leading dim ld.
__device__ __forceinline__ v16bf load_frag_a(const bf16_t* p, int ld, int row0, int k0) {
  int lane = threadIdx.x & 31;
  int half = lane >> 4;
  const bf16_t* base = p + (size_t)(row0 + (lane & 15)) * ld + k0 + 8 * half;
  v8bf c0 = *(const v8bf*)(base);
  v8bf c1 = *(const v8bf*)(base + 16);
  v16bf r;
#pragma unroll
  for (int i = 0; i < 8; ++i) { r[i] = c0[i]; r[i + 8] = c1[i]; }
  return r;
}

// B-fragment (32x16, KxN) with B[k][n] = p[n*ld + k].
__device__ __forceinline__ v16bf load_frag_b(const bf16_t* p, int ld, int col0, int k0) {
  int lane = threadIdx.x & 31;
  int half = lane >> 4;
  const bf16_t* base = p + (size_t)(col0 + (lane & 15)) * ld + k0 + 8 * half;
  v8bf c0 = *(const v8bf*)(base);
  v8bf c1 = *(const v8bf*)(base + 16);
  v16bf r;
#pragma unroll
  for (int i = 0; i < 8; ++i) { r[i] = c0[i]; r[i + 8] = c1[i]; }
  return r;
}

__device__ __forceinline__ void store_tile_f32(float* p, int ld, int row0, int col0, v8f c) {
  int lane = threadIdx.x & 31;
  int half = lane >> 4;
  int n = col0 + (lane & 15);
#pragma unroll
  for (int j = 0; j < 8; ++j)
    p[(size_t)(row0 + j + 8 * half) * ld + n] = c[j];
}

__device__ __forceinline__ void store_tile_bf16(bf16_t* p, int ld, int row0, int col0, v8f c) {
  int lane = threadIdx.x & 31;
  int half = lane >> 4;
  int n = col0 + (lane & 15);
#pragma unroll
  for (int j = 0; j < 8; ++j)
    p[(size_t)(row0 + j + 8 * half) * ld + n] = (bf16_t)c[j];
}

// ---------------------------------------------------------------------------
// fp32 -> bf16 conversion
// ---------------------------------------------------------------------------
__global__ void f32_to_bf16_kernel(const float* __restrict__ in,
                                   bf16_t* __restrict__ out, int n) {
  int i = blockIdx.x * blockDim.x + threadIdx.x;
  if (i < n) out[i] = (bf16_t)in[i];
}

// ---------------------------------------------------------------------------
// GEMM: out[m][n] = sum_k A[m][k] * W[n][k]   (y = x @ W^T)
// Block: 256 threads = 8 waves, tile 128(M) x 64(N); K-step 32.
// A/W tiles staged to LDS via async global->LDS; fragments via ds_load_b128.
// ---------------------------------------------------------------------------
template <bool OUT_F32>
__global__ void gemm_xwt_kernel(const bf16_t* __restrict__ A,
                                const bf16_t* __restrict__ W,
                                void* __restrict__ outp, int Kd) {
  __shared__ __align__(16) bf16_t Asm[128 * 32];   // 8KB
  __shared__ __align__(16) bf16_t Wsm[64 * 32];    // 4KB

  const int tid = threadIdx.x;
  const int wid = tid >> 5;
  const int wm = wid & 3;        // 4 waves along M
  const int wn = wid >> 2;       // 2 waves along N
  const int rowB = blockIdx.x * 128;
  const int colB = blockIdx.y * 64;
  const int row0 = wm * 32;      // within block tile
  const int col0 = wn * 32;

  v8f acc[2][2];
#pragma unroll
  for (int i = 0; i < 2; ++i)
#pragma unroll
    for (int j = 0; j < 2; ++j)
#pragma unroll
      for (int e = 0; e < 8; ++e) acc[i][j][e] = 0.0f;

  for (int k = 0; k < Kd; k += 32) {
    // stage A tile (128x32): 512 16B chunks; W tile (64x32): 256 chunks.
    for (int i = tid; i < 512; i += 256) {
      int r = i >> 2, c0 = (i & 3) * 8;
      async_copy16(&Asm[r * 32 + c0], &A[(size_t)(rowB + r) * Kd + k + c0]);
    }
    {
      int i = tid;   // 256 threads cover 256 chunks exactly
      int r = i >> 2, c0 = (i & 3) * 8;
      async_copy16(&Wsm[r * 32 + c0], &W[(size_t)(colB + r) * Kd + k + c0]);
    }
    async_copy_wait();
    __syncthreads();

    v16bf a0 = load_frag_a(Asm, 32, row0,      0);
    v16bf a1 = load_frag_a(Asm, 32, row0 + 16, 0);
    v16bf b0 = load_frag_b(Wsm, 32, col0,      0);
    v16bf b1 = load_frag_b(Wsm, 32, col0 + 16, 0);
    acc[0][0] = wmma_bf16(a0, b0, acc[0][0]);
    acc[0][1] = wmma_bf16(a0, b1, acc[0][1]);
    acc[1][0] = wmma_bf16(a1, b0, acc[1][0]);
    acc[1][1] = wmma_bf16(a1, b1, acc[1][1]);
    __syncthreads();
  }

#pragma unroll
  for (int i = 0; i < 2; ++i)
#pragma unroll
    for (int j = 0; j < 2; ++j) {
      if (OUT_F32)
        store_tile_f32((float*)outp, F_, rowB + row0 + 16 * i, colB + col0 + 16 * j, acc[i][j]);
      else
        store_tile_bf16((bf16_t*)outp, F_, rowB + row0 + 16 * i, colB + col0 + 16 * j, acc[i][j]);
    }
}

// ---------------------------------------------------------------------------
// Flash attention (causal, softclamped) per (b,h).
// Block: 128 threads = 4 waves; wave owns 16 query rows; K tile via TDM.
// ---------------------------------------------------------------------------
__global__ void attn_kernel(const bf16_t* __restrict__ Q,
                            const bf16_t* __restrict__ K,
                            const bf16_t* __restrict__ V,
                            bf16_t* __restrict__ O) {
  __shared__ __align__(16) bf16_t Kt[32 * 64];       // [key][dh] row-major
  __shared__ __align__(16) bf16_t Vt[64 * 32];       // [dh][key] (transposed)
  __shared__ __align__(16) bf16_t Ps[4][16 * 32];    // per-wave P staging

  const int tid  = threadIdx.x;
  const int lane = tid & 31;
  const int wid  = tid >> 5;
  const int half = lane >> 4;
  const int ln   = lane & 15;

  const int bh = blockIdx.y;
  const int b  = bh / H_;
  const int h  = bh % H_;
  const size_t headOff = (size_t)b * N_ * F_ + h * DH_;

  const bf16_t* Qh = Q + headOff;
  const bf16_t* Kh = K + headOff;
  const bf16_t* Vh = V + headOff;
  bf16_t*       Oh = O + headOff;

  const int row0w = blockIdx.x * 64 + wid * 16;   // wave's first query row

  v16bf qf[2];
#pragma unroll
  for (int ks = 0; ks < 2; ++ks)
    qf[ks] = load_frag_a(Qh, F_, row0w, 32 * ks);

  float m_run[8], l_run[8];
  v8f   o_acc[4];
#pragma unroll
  for (int j = 0; j < 8; ++j) { m_run[j] = -1e30f; l_run[j] = 0.0f; }
#pragma unroll
  for (int dt = 0; dt < 4; ++dt)
#pragma unroll
    for (int e = 0; e < 8; ++e) o_acc[dt][e] = 0.0f;

  const int ntiles = blockIdx.x * 2 + 2;

  for (int kt = 0; kt < ntiles; ++kt) {
    const int j0 = kt * 32;

    // K tile (32 keys x 64 dh, bf16) via one TDM descriptor from wave 0.
#if HAVE_TDM
    if (wid == 0)
      tdm_load_2d(Kt, Kh + (size_t)j0 * F_, 64u, 32u, (unsigned)F_);
#else
    for (int i = tid; i < 256; i += 128) {
      int r = i >> 2, c0 = (i & 3) * 8;
      async_copy16(&Kt[r * 64 + c0], &Kh[(size_t)(j0 + r) * F_ + c0]);
    }
#endif
    // V tile transposed: vector global load, scatter to LDS (overlaps TDM).
    for (int i = tid; i < 256; i += 128) {
      int r = i >> 2, c0 = (i & 3) * 8;
      v8bf vv = *(const v8bf*)&Vh[(size_t)(j0 + r) * F_ + c0];
#pragma unroll
      for (int e = 0; e < 8; ++e) Vt[(c0 + e) * 32 + r] = vv[e];
    }
#if HAVE_TDM
    if (wid == 0) tdm_wait();
#else
    async_copy_wait();
#endif
    __syncthreads();

    // S = Q * K^T
    v8f s[2];
#pragma unroll
    for (int nt = 0; nt < 2; ++nt) {
#pragma unroll
      for (int e = 0; e < 8; ++e) s[nt][e] = 0.0f;
#pragma unroll
      for (int ks = 0; ks < 2; ++ks) {
        v16bf kf = load_frag_b(Kt, 64, nt * 16, 32 * ks);
        s[nt] = wmma_bf16(qf[ks], kf, s[nt]);
      }
    }

    // softclamp + causal mask + online softmax; stage P into LDS (bf16).
    bf16_t* Psw = Ps[wid];
#pragma unroll
    for (int j = 0; j < 8; ++j) {
      const int rowg = row0w + j + 8 * half;
      float s0 = s[0][j] * SCALE_;
      float s1 = s[1][j] * SCALE_;
      s0 = CLAMP_ * tanhf(s0 * (1.0f / CLAMP_));
      s1 = CLAMP_ * tanhf(s1 * (1.0f / CLAMP_));
      if (j0 + ln > rowg)       s0 = -1e30f;
      if (j0 + 16 + ln > rowg)  s1 = -1e30f;

      float tmax = fmaxf(s0, s1);
#pragma unroll
      for (int off = 1; off < 16; off <<= 1)
        tmax = fmaxf(tmax, __shfl_xor(tmax, off, 32));

      float mold = m_run[j];
      float mnew = fmaxf(mold, tmax);
      float alpha = __expf(mold - mnew);
      float p0 = __expf(s0 - mnew);
      float p1 = __expf(s1 - mnew);
      float rs = p0 + p1;
#pragma unroll
      for (int off = 1; off < 16; off <<= 1)
        rs += __shfl_xor(rs, off, 32);

      l_run[j] = l_run[j] * alpha + rs;
      m_run[j] = mnew;
#pragma unroll
      for (int dt = 0; dt < 4; ++dt) o_acc[dt][j] *= alpha;

      const int rloc = j + 8 * half;
      Psw[rloc * 32 + ln]      = (bf16_t)p0;
      Psw[rloc * 32 + 16 + ln] = (bf16_t)p1;
    }
    asm volatile("s_wait_dscnt 0" ::: "memory");

    // O += P(16x32) * V(32x16 per dh tile)
    v16bf pa = load_frag_a(Psw, 32, 0, 0);
#pragma unroll
    for (int dt = 0; dt < 4; ++dt) {
      v16bf bv = load_frag_b(Vt, 32, dt * 16, 0);
      o_acc[dt] = wmma_bf16(pa, bv, o_acc[dt]);
    }
    __syncthreads();
  }

  float inv[8];
#pragma unroll
  for (int j = 0; j < 8; ++j) inv[j] = 1.0f / l_run[j];
#pragma unroll
  for (int dt = 0; dt < 4; ++dt) {
#pragma unroll
    for (int j = 0; j < 8; ++j) {
      int rowg = row0w + j + 8 * half;
      Oh[(size_t)rowg * F_ + dt * 16 + ln] = (bf16_t)(o_acc[dt][j] * inv[j]);
    }
  }
}

// ---------------------------------------------------------------------------
// Launcher
// ---------------------------------------------------------------------------
extern "C" void kernel_launch(void* const* d_in, const int* in_sizes, int n_in,
                              void* d_out, int out_size, void* d_ws, size_t ws_size,
                              hipStream_t stream) {
  const float* x  = (const float*)d_in[0];
  const float* Wq = (const float*)d_in[1];
  const float* Wk = (const float*)d_in[2];
  const float* Wv = (const float*)d_in[3];
  const float* Wo = (const float*)d_in[4];
  float* out = (float*)d_out;

  const size_t nX = (size_t)B_ * N_ * D_;   // 4,194,304
  const size_t nW = (size_t)F_ * D_;        // 1,048,576

  char* ws = (char*)d_ws;
  size_t off = 0;
  bf16_t* xb  = (bf16_t*)(ws + off); off += nX * sizeof(bf16_t);
  bf16_t* wqb = (bf16_t*)(ws + off); off += nW * sizeof(bf16_t);
  bf16_t* wkb = (bf16_t*)(ws + off); off += nW * sizeof(bf16_t);
  bf16_t* wvb = (bf16_t*)(ws + off); off += nW * sizeof(bf16_t);
  bf16_t* wob = (bf16_t*)(ws + off); off += nW * sizeof(bf16_t);
  bf16_t* Qb  = (bf16_t*)(ws + off); off += nX * sizeof(bf16_t);
  bf16_t* Kb  = (bf16_t*)(ws + off); off += nX * sizeof(bf16_t);
  bf16_t* Vb  = (bf16_t*)(ws + off); off += nX * sizeof(bf16_t);
  bf16_t* Ob  = (bf16_t*)(ws + off); off += nX * sizeof(bf16_t);

  f32_to_bf16_kernel<<<(int)((nX + 255) / 256), 256, 0, stream>>>(x, xb, (int)nX);
  f32_to_bf16_kernel<<<(int)((nW + 255) / 256), 256, 0, stream>>>(Wq, wqb, (int)nW);
  f32_to_bf16_kernel<<<(int)((nW + 255) / 256), 256, 0, stream>>>(Wk, wkb, (int)nW);
  f32_to_bf16_kernel<<<(int)((nW + 255) / 256), 256, 0, stream>>>(Wv, wvb, (int)nW);
  f32_to_bf16_kernel<<<(int)((nW + 255) / 256), 256, 0, stream>>>(Wo, wob, (int)nW);

  dim3 ggrid(4096 / 128, 1024 / 64);
  gemm_xwt_kernel<false><<<ggrid, 256, 0, stream>>>(xb, wqb, Qb, D_);
  gemm_xwt_kernel<false><<<ggrid, 256, 0, stream>>>(xb, wkb, Kb, D_);
  gemm_xwt_kernel<false><<<ggrid, 256, 0, stream>>>(xb, wvb, Vb, D_);

  dim3 agrid(N_ / 64, B_ * H_);
  attn_kernel<<<agrid, 128, 0, stream>>>(Qb, Kb, Vb, Ob);

  gemm_xwt_kernel<true><<<ggrid, 256, 0, stream>>>(Ob, wob, (void*)out, F_);
}
